// UpTransformer_1967095021883
// MI455X (gfx1250) — matrix operand order repaired and checked by reading, not compiled, not run on hardware
//
#include <hip/hip_runtime.h>
#include <hip/hip_bf16.h>

typedef __attribute__((ext_vector_type(16))) _Float16 v16h;
typedef __attribute__((ext_vector_type(8)))  float    v8f;

// ---------------- problem constants ----------------
constexpr int B_   = 4;
constexpr int N_   = 2048;
constexpr int CIN_ = 128;
constexpr int DIM_ = 64;
constexpr int KNN_ = 20;
constexpr int UP_  = 2;
constexpr int HID_ = 256;          // DIM*MULT
constexpr int COUT_= 128;
constexpr int NU_  = N_ * UP_;     // 4096
constexpr long S_  = (long)B_ * N_ * KNN_;   // 163840 samples
constexpr float EPS_ = 1e-5f;

// ---------------- WMMA fragment loaders ----------------
// A fragment 16x32 (MxK), W row-major [O x C]: element(m,k)=W[m*ld+k]
// lane: m = m0 + (lane&15); half h: k = c0 + 16*(lane>>4) + h
__device__ __forceinline__ v16h load_afrag(const float* __restrict__ W, int ld,
                                           int m0, int c0, int lane) {
  int m  = m0 + (lane & 15);
  int hi = lane >> 4;
  v16h a;
#pragma unroll
  for (int h = 0; h < 16; ++h) {
    int k = c0 + 16 * hi + h;
    a[h] = (_Float16)W[(long)m * ld + k];
  }
  return a;
}

// ---------------- generic point-wise conv (GEMM) via WMMA ----------------
// out(b,m,p) = relu?( W[m,:]*[in0;in1](:,p) + bias[m] ) + addsrc(b,m,p/res_ndiv)
// 4-wave workgroup, one 16-point tile per block. Activation panel (C x 16 fp32)
// is staged Global->LDS with CDNA5 async copies (ASYNCcnt), consumed by all waves.
__global__ void gemm_pw(float* __restrict__ out,
                        const float* __restrict__ in0, const float* __restrict__ in1,
                        const float* __restrict__ W,   const float* __restrict__ bias,
                        const float* __restrict__ addsrc,
                        int O, int C0, int C1, int Npts, int res_ndiv, int do_relu) {
  __shared__ __align__(16) float panel[256 * 16];    // [c][n], up to 16 KB
  int tid  = threadIdx.x;                            // 128 threads = 4 waves
  int lane = tid & 31, wv = tid >> 5;
  int n = lane & 15, hi = lane >> 4;
  long pt = (long)blockIdx.x * 16;                   // global point across batches
  int b  = (int)(pt / Npts);
  int np = (int)(pt % Npts);
  int C  = C0 + C1;

  // --- async DMA: C channel-rows of 16 fp32 (64 B) each into LDS ---
  for (int c = tid; c < C; c += 128) {
    const float* src = (c < C0)
        ? in0 + ((long)b * C0 + c) * Npts + np
        : in1 + ((long)b * C1 + (c - C0)) * Npts + np;
    unsigned ldsa = (unsigned)(unsigned long long)&panel[c * 16];
#pragma unroll
    for (int i = 0; i < 4; ++i) {
      asm volatile("global_load_async_to_lds_b128 %0, %1, off"
                   :: "v"(ldsa + 16u * i), "v"(src + 4 * i)
                   : "memory");
    }
  }
  asm volatile("s_wait_asynccnt 0x0" ::: "memory");
  __syncthreads();

  // --- each wave owns O/64 M-tiles over the shared panel ---
  int mt_per_wave = (O >> 4) >> 2;                   // O=128 -> 2, O=64 -> 1
  for (int mi = 0; mi < mt_per_wave; ++mi) {
    int m0 = (wv * mt_per_wave + mi) * 16;
    v8f acc = {};
    for (int c0 = 0; c0 < C; c0 += 32) {
      v16h a = load_afrag(W, C, m0, c0, lane);
      v16h bf;
#pragma unroll
      for (int h = 0; h < 16; ++h)
        bf[h] = (_Float16)panel[(c0 + 16 * hi + h) * 16 + n];
      acc = __builtin_amdgcn_wmma_f32_16x16x32_f16(false, a, false, bf, (short)0, acc, false, false);
    }
#pragma unroll
    for (int r = 0; r < 8; ++r) {
      int m = m0 + r + 8 * hi;
      float v = acc[r] + bias[m];
      if (do_relu) v = fmaxf(v, 0.f);
      if (addsrc) v += addsrc[((long)b * O + m) * (long)(Npts / res_ndiv) + (np + n) / res_ndiv];
      out[((long)b * O + m) * (long)Npts + np + n] = v;
    }
  }
}

// ---------------- KNN top-20 (stable, includes self) ----------------
__global__ void knn_kernel(const float* __restrict__ pos, int* __restrict__ idx) {
  __shared__ float sx[128], sy[128], sz[128];
  int t  = threadIdx.x;
  int gi = blockIdx.x * 128 + t;             // global point, < B*N
  int b  = gi / N_, n = gi % N_;
  const float* pb = pos + (long)b * 3 * N_;
  float xi = pb[n], yi = pb[N_ + n], zi = pb[2 * N_ + n];
  float dbest[KNN_]; int ibest[KNN_];
#pragma unroll
  for (int k = 0; k < KNN_; ++k) { dbest[k] = 3.4e38f; ibest[k] = 0; }
  for (int j0 = 0; j0 < N_; j0 += 128) {
    sx[t] = pb[j0 + t]; sy[t] = pb[N_ + j0 + t]; sz[t] = pb[2 * N_ + j0 + t];
    __syncthreads();
    for (int jj = 0; jj < 128; ++jj) {
      float dx = xi - sx[jj], dy = yi - sy[jj], dz = zi - sz[jj];
      float d = dx * dx + dy * dy + dz * dz;
      if (d < dbest[KNN_ - 1]) {
        float nd = d; int ni = j0 + jj;
#pragma unroll
        for (int k = 0; k < KNN_; ++k) {
          if (nd < dbest[k]) {
            float td = dbest[k]; int ti = ibest[k];
            dbest[k] = nd; ibest[k] = ni; nd = td; ni = ti;
          }
        }
      }
    }
    __syncthreads();
  }
#pragma unroll
  for (int k = 0; k < KNN_; ++k) idx[(long)gi * KNN_ + k] = ibest[k];
}

// ---------------- zero small stats region ----------------
__global__ void zero_kernel(float* __restrict__ p, int n) {
  int i = blockIdx.x * 256 + threadIdx.x;
  if (i < n) p[i] = 0.f;
}

// ---------------- pos_rel moment reduction (for analytic BN) ----------------
__global__ void pos_stats(const float* __restrict__ pos, const int* __restrict__ idx,
                          float* __restrict__ st /*9 floats*/) {
  __shared__ float acc[9];
  int t = threadIdx.x;
  if (t < 9) acc[t] = 0.f;
  __syncthreads();
  long sid = (long)blockIdx.x * 256 + t;     // < S_
  int b = (int)(sid / ((long)N_ * KNN_));
  int rem = (int)(sid % ((long)N_ * KNN_));
  int n = rem / KNN_;
  int j = idx[sid];
  const float* pb = pos + (long)b * 3 * N_;
  float dx = pb[n] - pb[j];
  float dy = pb[N_ + n] - pb[N_ + j];
  float dz = pb[2 * N_ + n] - pb[2 * N_ + j];
  atomicAdd(&acc[0], dx);      atomicAdd(&acc[1], dy);      atomicAdd(&acc[2], dz);
  atomicAdd(&acc[3], dx * dx); atomicAdd(&acc[4], dx * dy); atomicAdd(&acc[5], dx * dz);
  atomicAdd(&acc[6], dy * dy); atomicAdd(&acc[7], dy * dz); atomicAdd(&acc[8], dz * dz);
  __syncthreads();
  if (t < 9) atomicAdd(&st[t], acc[t]);
}

// fold pos-BN (train mode) into per-channel scale/shift from 3x3 input moments
__global__ void pos_bn_prep(const float* __restrict__ st,
                            const float* __restrict__ p1w, const float* __restrict__ p1b,
                            const float* __restrict__ g,   const float* __restrict__ be,
                            float* __restrict__ scale, float* __restrict__ shift) {
  int c = threadIdx.x;  // 64
  const float S = (float)S_;
  float mu0 = st[0] / S, mu1 = st[1] / S, mu2 = st[2] / S;
  float M00 = st[3] / S, M01 = st[4] / S, M02 = st[5] / S;
  float M11 = st[6] / S, M12 = st[7] / S, M22 = st[8] / S;
  float w0 = p1w[c * 3], w1 = p1w[c * 3 + 1], w2 = p1w[c * 3 + 2], bb = p1b[c];
  float wmu  = w0 * mu0 + w1 * mu1 + w2 * mu2;
  float mean = wmu + bb;
  float wMw = w0 * (w0 * M00 + w1 * M01 + w2 * M02)
            + w1 * (w0 * M01 + w1 * M11 + w2 * M12)
            + w2 * (w0 * M02 + w1 * M12 + w2 * M22);
  float Ey2 = wMw + 2.f * bb * wmu + bb * bb;
  float var = Ey2 - mean * mean;
  float sc  = g[c] * rsqrtf(var + EPS_);
  scale[c] = sc;
  shift[c] = be[c] - mean * sc;
}

// ---------------- build s = qk_rel+pe+uf_rel and w = v_g+pe+uf_rel (f16, [p][64]) ----------------
__global__ void build_sw(const float* __restrict__ pos, const int* __restrict__ idx,
                         const float* __restrict__ kp, const float* __restrict__ qp,
                         const float* __restrict__ vp, const float* __restrict__ up,
                         const float* __restrict__ p1w, const float* __restrict__ p1b,
                         const float* __restrict__ psc, const float* __restrict__ psh,
                         const float* __restrict__ p2w, const float* __restrict__ p2b,
                         _Float16* __restrict__ s16, _Float16* __restrict__ w16) {
  long sid = (long)blockIdx.x * 64 + threadIdx.x;   // < S_
  int b = (int)(sid / ((long)N_ * KNN_));
  int rem = (int)(sid % ((long)N_ * KNN_));
  int n = rem / KNN_;
  int j = idx[sid];
  const float* pb = pos + (long)b * 3 * N_;
  float pr0 = pb[n] - pb[j];
  float pr1 = pb[N_ + n] - pb[N_ + j];
  float pr2 = pb[2 * N_ + n] - pb[2 * N_ + j];
  float z[64];
#pragma unroll
  for (int c = 0; c < 64; ++c) {
    float y = p1w[c * 3] * pr0 + p1w[c * 3 + 1] * pr1 + p1w[c * 3 + 2] * pr2 + p1b[c];
    z[c] = fmaxf(y * psc[c] + psh[c], 0.f);
  }
  long cb = (long)b * DIM_ * N_;
  for (int c = 0; c < DIM_; ++c) {
    float pe = p2b[c];
#pragma unroll
    for (int jj = 0; jj < 64; ++jj) pe += p2w[c * 64 + jj] * z[jj];
    long rn = cb + (long)c * N_ + n;
    long rj = cb + (long)c * N_ + j;
    float urel = up[rn] - up[rj];
    float sval = qp[rn] - kp[rj] + pe + urel;
    float wval = vp[rj] + pe + urel;
    s16[sid * DIM_ + c] = (_Float16)sval;
    w16[sid * DIM_ + c] = (_Float16)wval;
  }
}

// ---------------- s second-moment reduction (64 + 64x64) ----------------
__global__ void s_stats(const _Float16* __restrict__ s16,
                        float* __restrict__ Ss /*64*/, float* __restrict__ Sss /*4096*/) {
  __shared__ _Float16 sch[128 * 64];   // 128 samples x 64 ch, 16 KB
  int t = threadIdx.x;                 // 256
  long base = (long)blockIdx.x * 128 * 64;
  for (int i = t; i < 128 * 64; i += 256) sch[i] = s16[base + i];
  __syncthreads();
  int i  = t >> 2;            // 0..63
  int j0 = (t & 3) * 16;      // 0,16,32,48
  float acc[16];
#pragma unroll
  for (int q = 0; q < 16; ++q) acc[q] = 0.f;
  for (int ss = 0; ss < 128; ++ss) {
    float ai = (float)sch[ss * 64 + i];
#pragma unroll
    for (int q = 0; q < 16; ++q) acc[q] += ai * (float)sch[ss * 64 + j0 + q];
  }
#pragma unroll
  for (int q = 0; q < 16; ++q) atomicAdd(&Sss[i * 64 + j0 + q], acc[q]);
  if (t < 64) {
    float m = 0.f;
    for (int ss = 0; ss < 128; ++ss) m += (float)sch[ss * 64 + t];
    atomicAdd(&Ss[t], m);
  }
}

// fold attn-BN into per-channel scale/shift from E[s], E[ss^T]
__global__ void attn_bn_prep(const float* __restrict__ Ss, const float* __restrict__ Sss,
                             const float* __restrict__ a1w, const float* __restrict__ a1b,
                             const float* __restrict__ g,   const float* __restrict__ be,
                             float* __restrict__ sc2, float* __restrict__ sh2) {
  int c = threadIdx.x;   // 256
  const float S = (float)S_;
  const float* a = a1w + c * 64;
  float bb = a1b[c];
  float mdot = 0.f;
#pragma unroll
  for (int i = 0; i < 64; ++i) mdot += a[i] * (Ss[i] / S);
  float mean = mdot + bb;
  float wMw = 0.f;
  for (int i = 0; i < 64; ++i) {
    float row = 0.f;
#pragma unroll
    for (int j = 0; j < 64; ++j) row += a[j] * (Sss[i * 64 + j] / S);
    wMw += a[i] * row;
  }
  float Ey2 = wMw + 2.f * bb * mdot + bb * bb;
  float var = Ey2 - mean * mean;
  float s = g[c] * rsqrtf(var + EPS_);
  sc2[c] = s;
  sh2[c] = be[c] - mean * s;
}

// ---------------- fused attention: a1 -> BN/ReLU -> ConvT -> softmax -> aggregate ----------------
// 1 wave / block, 4 points (80 samples = 5 sample-tiles). h and attn live only in LDS.
__global__ void attn_fused(const _Float16* __restrict__ s16, const _Float16* __restrict__ w16,
                           const float* __restrict__ a1w,
                           const float* __restrict__ sc2, const float* __restrict__ sh2,
                           const float* __restrict__ aTw, const float* __restrict__ aTb,
                           float* __restrict__ agg) {
  __shared__ __align__(32) _Float16 hst[16 * HID_];   // [sample16][ch256]  8 KB
  __shared__ float tl[128 * 80];                      // [o*2+r][sample80] 40 KB
  int lane = threadIdx.x;
  int n = lane & 15, hi = lane >> 4;
  long blk = blockIdx.x;             // 2048 blocks
  long p0  = blk * 80;               // global sample base
  int  g0  = (int)(blk * 4);         // global point base
  int  b   = g0 / N_;
  int  nl0 = g0 % N_;

  for (int st = 0; st < 5; ++st) {
    long sb = p0 + (long)st * 16;
    // GEMM1: h = relu(bn(a1 * s)), s f16 sample-major -> contiguous B-fragments
    for (int mt = 0; mt < 16; ++mt) {
      v8f acc = {};
#pragma unroll
      for (int ks = 0; ks < 2; ++ks) {
        v16h a  = load_afrag(a1w, DIM_, mt * 16, ks * 32, lane);
        v16h bf = *(const v16h*)(s16 + (sb + n) * DIM_ + ks * 32 + hi * 16);
        acc = __builtin_amdgcn_wmma_f32_16x16x32_f16(false, a, false, bf, (short)0, acc, false, false);
      }
#pragma unroll
      for (int r = 0; r < 8; ++r) {
        int ch = mt * 16 + r + 8 * hi;
        float hv = fmaxf(acc[r] * sc2[ch] + sh2[ch], 0.f);
        hst[n * HID_ + ch] = (_Float16)hv;
      }
    }
    __syncthreads();
    // GEMM2: t[o*2+r] = Wt2 * h, Wt2[or][c] = aT_w[c*128 + or]
    for (int mt = 0; mt < 8; ++mt) {
      v8f acc = {};
#pragma unroll
      for (int ks = 0; ks < 8; ++ks) {
        v16h a;
        int m = lane & 15;
#pragma unroll
        for (int h = 0; h < 16; ++h) {
          int kk = ks * 32 + 16 * hi + h;
          a[h] = (_Float16)aTw[(long)kk * 128 + mt * 16 + m];
        }
        v16h bf = *(const v16h*)(&hst[n * HID_ + ks * 32 + hi * 16]);
        acc = __builtin_amdgcn_wmma_f32_16x16x32_f16(false, a, false, bf, (short)0, acc, false, false);
      }
#pragma unroll
      for (int r = 0; r < 8; ++r) {
        int orr = mt * 16 + r + 8 * hi;                 // o*2+r
        tl[orr * 80 + st * 16 + n] = acc[r] + aTb[orr >> 1];
      }
    }
    __syncthreads();
  }
  // softmax over K=20 per (or, local point), then weighted aggregation with w
  for (int row = lane; row < 128 * 4; row += 32) {
    int orr = row >> 2, nl = row & 3;
    int o = orr >> 1, r = orr & 1;
    const float* tv = &tl[orr * 80 + nl * 20];
    float mx = -3.4e38f;
#pragma unroll
    for (int k = 0; k < KNN_; ++k) mx = fmaxf(mx, tv[k]);
    float e[KNN_], sum = 0.f;
#pragma unroll
    for (int k = 0; k < KNN_; ++k) { e[k] = __expf(tv[k] - mx); sum += e[k]; }
    float inv = 1.f / sum;
    const _Float16* wv = w16 + (p0 + (long)nl * KNN_) * DIM_ + o;
    float a = 0.f;
#pragma unroll
    for (int k = 0; k < KNN_; ++k) a += e[k] * inv * (float)wv[(long)k * DIM_];
    int ng = nl0 + nl;
    agg[((long)b * DIM_ + o) * NU_ + ng * 2 + r] = a;
  }
}

// ==================== host side ====================
extern "C" void kernel_launch(void* const* d_in, const int* in_sizes, int n_in,
                              void* d_out, int out_size, void* d_ws, size_t ws_size,
                              hipStream_t stream) {
  (void)in_sizes; (void)n_in; (void)out_size; (void)ws_size;
  const float* pos    = (const float*)d_in[0];
  const float* key    = (const float*)d_in[1];
  const float* query  = (const float*)d_in[2];
  const float* upfeat = (const float*)d_in[3];
  const float* v1_w = (const float*)d_in[4];  const float* v1_b = (const float*)d_in[5];
  const float* v2_w = (const float*)d_in[6];  const float* v2_b = (const float*)d_in[7];
  const float* vs_w = (const float*)d_in[8];  const float* vs_b = (const float*)d_in[9];
  const float* k_w  = (const float*)d_in[10]; const float* k_b  = (const float*)d_in[11];
  const float* q_w  = (const float*)d_in[12]; const float* q_b  = (const float*)d_in[13];
  const float* v_w  = (const float*)d_in[14]; const float* v_b  = (const float*)d_in[15];
  const float* u_w  = (const float*)d_in[16]; const float* u_b  = (const float*)d_in[17];
  const float* p1_w = (const float*)d_in[18]; const float* p1_b = (const float*)d_in[19];
  const float* pbn_g= (const float*)d_in[20]; const float* pbn_b= (const float*)d_in[21];
  const float* p2_w = (const float*)d_in[22]; const float* p2_b = (const float*)d_in[23];
  const float* a1_w = (const float*)d_in[24]; const float* a1_b = (const float*)d_in[25];
  const float* abn_g= (const float*)d_in[26]; const float* abn_b= (const float*)d_in[27];
  const float* aT_w = (const float*)d_in[28]; const float* aT_b = (const float*)d_in[29];
  const float* e_w  = (const float*)d_in[30]; const float* e_b  = (const float*)d_in[31];
  float* out = (float*)d_out;

  // ---- workspace layout ----
  char* ws = (char*)d_ws;
  size_t off = 0;
  auto take = [&](size_t bytes) { void* p = ws + off; off += (bytes + 255) & ~(size_t)255; return p; };
  float* value = (float*)take((size_t)B_ * CIN_ * N_ * 4);     // 4 MB
  float* h1    = (float*)take((size_t)B_ * CIN_ * N_ * 4);     // 4 MB
  float* kp    = (float*)take((size_t)B_ * DIM_ * N_ * 4);
  float* qp    = (float*)take((size_t)B_ * DIM_ * N_ * 4);
  float* vp    = (float*)take((size_t)B_ * DIM_ * N_ * 4);
  float* up    = (float*)take((size_t)B_ * DIM_ * N_ * 4);
  float* agg   = (float*)take((size_t)B_ * DIM_ * NU_ * 4);    // 4 MB
  float* stats = (float*)take(32768);                          // atomics + folded BN params
  float* st9  = stats;            // 9  (zeroed)
  float* Ss   = stats + 16;       // 64 (zeroed)
  float* Sss  = stats + 80;       // 4096 (zeroed)
  float* psc  = stats + 4176;     // 64
  float* psh  = stats + 4240;     // 64
  float* sc2  = stats + 4304;     // 256
  float* sh2  = stats + 4560;     // 256
  int*      idx = (int*)take((size_t)S_ * 4);
  _Float16* s16 = (_Float16*)take((size_t)S_ * DIM_ * 2);      // 20 MB
  _Float16* w16 = (_Float16*)take((size_t)S_ * DIM_ * 2);      // 20 MB

  // 0) zero atomic accumulators
  zero_kernel<<<dim3((4176 + 255) / 256), dim3(256), 0, stream>>>(stats, 4176);

  // 1) value = v2(relu(v1([key;query]))) + vs([key;query])
  gemm_pw<<<dim3(B_ * N_ / 16), dim3(128), 0, stream>>>(
      h1, key, query, v1_w, v1_b, nullptr, CIN_, CIN_, CIN_, N_, 1, 1);
  gemm_pw<<<dim3(B_ * N_ / 16), dim3(128), 0, stream>>>(
      value, key, query, vs_w, vs_b, nullptr, CIN_, CIN_, CIN_, N_, 1, 0);
  gemm_pw<<<dim3(B_ * N_ / 16), dim3(128), 0, stream>>>(
      value, h1, nullptr, v2_w, v2_b, value, CIN_, CIN_, 0, N_, 1, 0);

  // 2) k/q/v/u projections (64 x 128)
  gemm_pw<<<dim3(B_ * N_ / 16), dim3(128), 0, stream>>>(
      kp, key, nullptr, k_w, k_b, nullptr, DIM_, CIN_, 0, N_, 1, 0);
  gemm_pw<<<dim3(B_ * N_ / 16), dim3(128), 0, stream>>>(
      qp, query, nullptr, q_w, q_b, nullptr, DIM_, CIN_, 0, N_, 1, 0);
  gemm_pw<<<dim3(B_ * N_ / 16), dim3(128), 0, stream>>>(
      vp, value, nullptr, v_w, v_b, nullptr, DIM_, CIN_, 0, N_, 1, 0);
  gemm_pw<<<dim3(B_ * N_ / 16), dim3(128), 0, stream>>>(
      up, upfeat, nullptr, u_w, u_b, nullptr, DIM_, CIN_, 0, N_, 1, 0);

  // 3) KNN top-20
  knn_kernel<<<dim3(B_ * N_ / 128), dim3(128), 0, stream>>>(pos, idx);

  // 4) pos-BN folded via 3x3 input moments
  pos_stats<<<dim3((unsigned)(S_ / 256)), dim3(256), 0, stream>>>(pos, idx, st9);
  pos_bn_prep<<<dim3(1), dim3(64), 0, stream>>>(st9, p1_w, p1_b, pbn_g, pbn_b, psc, psh);

  // 5) build s (attn input) and w (value-side) in f16 sample-major
  build_sw<<<dim3((unsigned)(S_ / 64)), dim3(64), 0, stream>>>(
      pos, idx, kp, qp, vp, up, p1_w, p1_b, psc, psh, p2_w, p2_b, s16, w16);

  // 6) attn-BN folded via 64x64 moments of s
  s_stats<<<dim3((unsigned)(S_ / 128)), dim3(256), 0, stream>>>(s16, Ss, Sss);
  attn_bn_prep<<<dim3(1), dim3(256), 0, stream>>>(Ss, Sss, a1_w, a1_b, abn_g, abn_b, sc2, sh2);

  // 7) fused attention MLP + softmax + aggregation
  attn_fused<<<dim3((unsigned)(S_ / 80)), dim3(32), 0, stream>>>(
      s16, w16, a1_w, sc2, sh2, aT_w, aT_b, agg);

  // 8) conv_end + upsampled residual
  gemm_pw<<<dim3(B_ * NU_ / 16), dim3(128), 0, stream>>>(
      out, agg, nullptr, e_w, e_b, value, COUT_, DIM_, 0, NU_, UP_, 0);
}
